// BertSelfAttention1D_3805341024745
// MI455X (gfx1250) — matrix operand order repaired and checked by reading, not compiled, not run on hardware
//
#include <hip/hip_runtime.h>
#include <hip/hip_bf16.h>

typedef __attribute__((ext_vector_type(16))) __bf16 v16bf;
typedef __attribute__((ext_vector_type(8)))  float  v8f;

#define B_    4
#define S_    2048
#define H_    1024
#define NH_   16
#define D_    64
#define BS_   (B_ * S_)      // 8192
#define NQKV_ (3 * H_)       // 3072

// ---------- helpers ----------

__device__ __forceinline__ unsigned short f2bf(float f) {
  unsigned u = __float_as_uint(f);
  unsigned r = (u + 0x7FFFu + ((u >> 16) & 1u)) >> 16;   // round-to-nearest-even
  return (unsigned short)r;
}

union Frag16 { unsigned u[8]; uint4 q[2]; v16bf v; };
union FragC  { float f[8]; v8f v; };

__device__ __forceinline__ v8f wmma_bf16(v16bf a, v16bf b, v8f c) {
  return __builtin_amdgcn_wmma_f32_16x16x32_bf16(false, a, false, b, (short)0, c,
                                                 false, false);
}

// A-matrix fragment (16x32 bf16), row-major source, 16B-aligned chunks.
// lane: M = lane&15, g = lane>>4; VGPR v: K = 16g + 2v (+1)
__device__ __forceinline__ v16bf load_a_frag(const unsigned short* rowptr, int kb, int lane) {
  int g = lane >> 4;
  Frag16 fr;
  fr.q[0] = *(const uint4*)(rowptr + kb + 16 * g);
  fr.q[1] = *(const uint4*)(rowptr + kb + 16 * g + 8);
  return fr.v;
}

// B-matrix fragment (32x16 bf16) from Bt stored [N,K] row-major.
// lane: N = lane&15, g = lane>>4; VGPR v: K = 16g + 2v (+1)
__device__ __forceinline__ v16bf load_b_frag(const unsigned short* rowptr, int kb, int lane) {
  int g = lane >> 4;
  Frag16 fr;
  fr.q[0] = *(const uint4*)(rowptr + kb + 16 * g);
  fr.q[1] = *(const uint4*)(rowptr + kb + 16 * g + 8);
  return fr.v;
}

// gfx1250 async copy: 16 bytes global -> LDS per lane, tracked by ASYNCcnt.
__device__ __forceinline__ void async_b128(unsigned lds_off, const unsigned short* g) {
  asm volatile("global_load_async_to_lds_b128 %0, %1, off"
               :: "v"(lds_off), "v"((unsigned long long)(size_t)g)
               : "memory");
}
__device__ __forceinline__ void wait_async0() {
  asm volatile("s_wait_asynccnt 0x0" ::: "memory");
}
__device__ __forceinline__ void wait_async1() {
  asm volatile("s_wait_asynccnt 0x1" ::: "memory");
}

// Block-cooperative 128(M) x 64(N) x H_ GEMM.
// B panel K-slices (64x32 bf16 = 4KB) double-buffered in LDS, filled with
// GLOBAL_LOAD_ASYNC_TO_LDS_B128 (ASYNCcnt); 8 waves share the panel and each
// computes a 16x64 strip with A fragments double-buffered in registers.
__device__ __forceinline__ void block_gemm_128x64(const unsigned short* __restrict__ Arow,
                                                  const unsigned short* __restrict__ Bpanel, // Wt + n0*H_
                                                  unsigned short (&ldsB)[2][64][32],
                                                  int lane, FragC acc[4]) {
  const int NSTEPS = H_ / 32;                    // 32
  int tid = threadIdx.x;
  int bn = tid >> 2;                             // 0..63 : B row within panel
  int bk = (tid & 3) * 8;                        // 0,8,16,24 : K offset (8 elems = 16B)
  int r = lane & 15, g = lane >> 4;
  const unsigned short* gsrc = Bpanel + (size_t)bn * H_ + bk;
  unsigned lds_off[2];
  lds_off[0] = (unsigned)(size_t)&ldsB[0][bn][bk];
  lds_off[1] = (unsigned)(size_t)&ldsB[1][bn][bk];

  // prologue: start filling buffer 0 with K-slice 0
  async_b128(lds_off[0], gsrc);
  v16bf a0 = load_a_frag(Arow, 0, lane);
  v16bf a1 = a0;

  for (int i = 0; i < NSTEPS; i++) {
    int kb = i * 32;
    __syncthreads();                             // all reads of buf[(i+1)&1] done
    if (i + 1 < NSTEPS) {
      async_b128(lds_off[(i + 1) & 1], gsrc + kb + 32);
      a1 = load_a_frag(Arow, kb + 32, lane);
      wait_async1();                             // buf[i&1] landed; next fill in flight
    } else {
      wait_async0();
    }
    __syncthreads();                             // every wave's portion visible

    const unsigned short(&bbuf)[64][32] = ldsB[i & 1];
#pragma unroll
    for (int t = 0; t < 4; t++) {
      Frag16 fb;
      fb.q[0] = *(const uint4*)&bbuf[16 * t + r][16 * g];
      fb.q[1] = *(const uint4*)&bbuf[16 * t + r][16 * g + 8];
      acc[t].v = wmma_bf16(a0, fb.v, acc[t].v);
    }
    a0 = a1;
  }
}

// ---------- kernel 1: f32 -> bf16 ----------

__global__ void __launch_bounds__(256) k_f32_to_bf16(const float* __restrict__ in,
                                                     unsigned short* __restrict__ out,
                                                     int n) {
  int i = (blockIdx.x * 256 + threadIdx.x) * 4;
  if (i + 3 < n) {
    float4 f = *(const float4*)(in + i);
    out[i + 0] = f2bf(f.x);
    out[i + 1] = f2bf(f.y);
    out[i + 2] = f2bf(f.z);
    out[i + 3] = f2bf(f.w);
  }
}

// ---------- kernel 2: transpose f32 [K,N] -> bf16 [N,K] ----------

__global__ void __launch_bounds__(256) k_transpose_bf16(const float* __restrict__ W,
                                                        unsigned short* __restrict__ Wt,
                                                        int K, int N) {
  __shared__ float tile[32][33];
  int tx = threadIdx.x & 31;
  int ty = threadIdx.x >> 5;       // 0..7
#pragma unroll
  for (int i = 0; i < 4; i++) {
    int k = blockIdx.y * 32 + ty + i * 8;
    int n = blockIdx.x * 32 + tx;
    tile[ty + i * 8][tx] = W[(size_t)k * N + n];
  }
  __syncthreads();
#pragma unroll
  for (int i = 0; i < 4; i++) {
    int n = blockIdx.x * 32 + ty + i * 8;
    int k = blockIdx.y * 32 + tx;
    Wt[(size_t)n * K + k] = f2bf(tile[tx][ty + i * 8]);
  }
}

// ---------- kernel 3: QKV GEMM (WMMA + async LDS pipeline) ----------
// C = Xb[BS,H] @ Wqkv + bias, scattered to Q/K/V [B,NH,S,D] bf16.
// grid: (BS/128) * (NQKV/64) blocks; block = 128x64 tile.

__global__ void __launch_bounds__(256) k_gemm_qkv(const unsigned short* __restrict__ Xb,
                                                  const unsigned short* __restrict__ Wt,   // [3H,H]
                                                  const float* __restrict__ bias,
                                                  unsigned short* __restrict__ Q,
                                                  unsigned short* __restrict__ Kd,
                                                  unsigned short* __restrict__ V) {
  __shared__ unsigned short ldsB[2][64][32];
  int lane = threadIdx.x & 31;
  int wave = threadIdx.x >> 5;
  const int ntn = NQKV_ / 64;                   // 48
  int m0 = (blockIdx.x / ntn) * 128 + wave * 16;
  int n0 = (blockIdx.x % ntn) * 64;
  int r = lane & 15, g = lane >> 4;

  FragC acc[4];
#pragma unroll
  for (int t = 0; t < 4; t++)
#pragma unroll
    for (int v = 0; v < 8; v++) acc[t].f[v] = 0.f;

  block_gemm_128x64(Xb + (size_t)(m0 + r) * H_, Wt + (size_t)n0 * H_, ldsB, lane, acc);

#pragma unroll
  for (int t = 0; t < 4; t++) {
    int n = n0 + 16 * t + r;                    // output column
    int h = n / 192;
    int rem = n - h * 192;
    int part = rem >> 6;                        // 0=q 1=k 2=v
    int dd = rem & 63;
    unsigned short* dst = (part == 0) ? Q : (part == 1) ? Kd : V;
    float bv = bias[n];
#pragma unroll
    for (int v = 0; v < 8; v++) {
      int m = m0 + v + 8 * g;                   // global row = b*S + s
      int bb = m >> 11;                         // /2048
      int s = m & (S_ - 1);
      size_t idx = ((size_t)(bb * NH_ + h) * S_ + s) * D_ + dd;
      dst[idx] = f2bf(acc[t].f[v] + bv);
    }
  }
}

// ---------- kernel 4: flash attention (WMMA, online softmax) ----------
// block = (b, h, 128 q rows); 8 waves x 16 q rows; key blocks of 32.

__global__ void __launch_bounds__(256) k_attention(const unsigned short* __restrict__ Q,
                                                   const unsigned short* __restrict__ Km,
                                                   const unsigned short* __restrict__ V,
                                                   const float* __restrict__ mask,   // [B,S]
                                                   unsigned short* __restrict__ ctx) // [BS,H] bf16
{
  __shared__ unsigned short Vt[D_][34];          // V tile transposed [d][key], padded
  __shared__ unsigned short Pl[8][16 * 34];      // per-wave P staging, row stride 34

  int bid = blockIdx.x;
  int qb = (bid & 15) * 128;                     // S_/128 = 16 q blocks
  int h = (bid >> 4) & (NH_ - 1);
  int b = bid >> 8;
  int lane = threadIdx.x & 31;
  int wave = threadIdx.x >> 5;
  int r = lane & 15, g = lane >> 4;

  const size_t headOff = (size_t)(b * NH_ + h) * S_ * D_;
  const unsigned short* Qh = Q + headOff;
  const unsigned short* Kh = Km + headOff;
  const unsigned short* Vh = V + headOff;

  int q0 = qb + wave * 16;
  const unsigned short* qrow = Qh + (size_t)(q0 + r) * D_;
  v16bf qf0 = load_a_frag(qrow, 0, lane);
  v16bf qf1 = load_a_frag(qrow, 32, lane);

  FragC o[4];
#pragma unroll
  for (int t = 0; t < 4; t++)
#pragma unroll
    for (int v = 0; v < 8; v++) o[t].f[v] = 0.f;
  float mrun[8], l[8];
#pragma unroll
  for (int v = 0; v < 8; v++) { mrun[v] = -1e30f; l[v] = 0.f; }

  const float sc = 0.125f;                       // 1/sqrt(64)
  unsigned short* Pw = &Pl[wave][0];
  int vkey = threadIdx.x >> 3;                   // 0..31 (coop V load)
  int vd0 = (threadIdx.x & 7) * 8;

  for (int kb = 0; kb < S_; kb += 32) {
    // Issue the V-tile load FIRST, then the K fragments.  Loads complete in
    // order, so waiting for the V data leaves the K fragments in flight
    // across the barrier + LDS transpose, overlapping their latency.
    uint4 vdat = *(const uint4*)(Vh + (size_t)(kb + vkey) * D_ + vd0);
    const unsigned short* K0 = Kh + (size_t)(kb + r) * D_;
    const unsigned short* K1 = Kh + (size_t)(kb + 16 + r) * D_;
    v16bf kf00 = load_b_frag(K0, 0, lane);
    v16bf kf01 = load_b_frag(K0, 32, lane);
    v16bf kf10 = load_b_frag(K1, 0, lane);
    v16bf kf11 = load_b_frag(K1, 32, lane);

    __syncthreads();
    {
      const unsigned short* e = (const unsigned short*)&vdat;
#pragma unroll
      for (int i = 0; i < 8; i++) Vt[vd0 + i][vkey] = e[i];
    }
    __syncthreads();

    // scores: 16q x 32keys as two C fragments (contraction d=64 -> 2 WMMA each)
    FragC s0, s1;
#pragma unroll
    for (int v = 0; v < 8; v++) { s0.f[v] = 0.f; s1.f[v] = 0.f; }
    s0.v = wmma_bf16(qf0, kf00, s0.v);
    s0.v = wmma_bf16(qf1, kf01, s0.v);
    s1.v = wmma_bf16(qf0, kf10, s1.v);
    s1.v = wmma_bf16(qf1, kf11, s1.v);

    float mk0 = mask[b * S_ + kb + r];
    float mk1 = mask[b * S_ + kb + 16 + r];
    float bmax[8];
#pragma unroll
    for (int v = 0; v < 8; v++) {
      s0.f[v] = s0.f[v] * sc + mk0;
      s1.f[v] = s1.f[v] * sc + mk1;
      bmax[v] = fmaxf(s0.f[v], s1.f[v]);
    }
#pragma unroll
    for (int mm = 1; mm < 16; mm <<= 1)
#pragma unroll
      for (int v = 0; v < 8; v++)
        bmax[v] = fmaxf(bmax[v], __shfl_xor(bmax[v], mm, 32));

    float rs[8];
#pragma unroll
    for (int v = 0; v < 8; v++) {
      float mnew = fmaxf(mrun[v], bmax[v]);
      float corr = __expf(mrun[v] - mnew);
      mrun[v] = mnew;
      float p0 = __expf(s0.f[v] - mnew);
      float p1 = __expf(s1.f[v] - mnew);
      Pw[(v + 8 * g) * 34 + r] = f2bf(p0);
      Pw[(v + 8 * g) * 34 + 16 + r] = f2bf(p1);
      rs[v] = p0 + p1;
      l[v] *= corr;
#pragma unroll
      for (int t = 0; t < 4; t++) o[t].f[v] *= corr;
    }
#pragma unroll
    for (int mm = 1; mm < 16; mm <<= 1)
#pragma unroll
      for (int v = 0; v < 8; v++) rs[v] += __shfl_xor(rs[v], mm, 32);
#pragma unroll
    for (int v = 0; v < 8; v++) l[v] += rs[v];

    // reload P in A-fragment layout (dword reads, row stride 34)
    Frag16 pa;
    const unsigned short* prow = Pw + r * 34;
#pragma unroll
  for (int v = 0; v < 4; v++) pa.u[v] = *(const unsigned*)(prow + 16 * g + 2 * v);
#pragma unroll
  for (int v = 0; v < 4; v++) pa.u[4 + v] = *(const unsigned*)(prow + 16 * g + 8 + 2 * v);

    // P @ V : 4 d-tiles, B fragments from transposed V in LDS
#pragma unroll
    for (int t = 0; t < 4; t++) {
      Frag16 fb;
      const unsigned short* vrow = &Vt[16 * t + r][0];
#pragma unroll
      for (int v = 0; v < 8; v++) fb.u[v] = *(const unsigned*)(vrow + 16 * g + 2 * v);
      o[t].v = wmma_bf16(pa.v, fb.v, o[t].v);
    }
  }

  // normalize and store ctx [B,S,H] with H index = h*64 + d
#pragma unroll
  for (int t = 0; t < 4; t++) {
#pragma unroll
    for (int v = 0; v < 8; v++) {
      int s = q0 + v + 8 * g;
      size_t idx = ((size_t)(b * S_) + s) * H_ + h * D_ + 16 * t + r;
      ctx[idx] = f2bf(o[t].f[v] / l[v]);
    }
  }
}

// ---------- kernel 5: dense GEMM (WMMA + async LDS pipeline) ----------

__global__ void __launch_bounds__(256) k_gemm_dense(const unsigned short* __restrict__ Cb,  // [BS,H]
                                                    const unsigned short* __restrict__ Wt,  // [H,H] (N,K)
                                                    const float* __restrict__ bias,
                                                    float* __restrict__ Out) {              // [BS,H]
  __shared__ unsigned short ldsB[2][64][32];
  int lane = threadIdx.x & 31;
  int wave = threadIdx.x >> 5;
  const int ntn = H_ / 64;                       // 16
  int m0 = (blockIdx.x / ntn) * 128 + wave * 16;
  int n0 = (blockIdx.x % ntn) * 64;
  int r = lane & 15, g = lane >> 4;

  FragC acc[4];
#pragma unroll
  for (int t = 0; t < 4; t++)
#pragma unroll
    for (int v = 0; v < 8; v++) acc[t].f[v] = 0.f;

  block_gemm_128x64(Cb + (size_t)(m0 + r) * H_, Wt + (size_t)n0 * H_, ldsB, lane, acc);

#pragma unroll
  for (int t = 0; t < 4; t++) {
    int n = n0 + 16 * t + r;
    float bv = bias[n];
#pragma unroll
    for (int v = 0; v < 8; v++)
      Out[(size_t)(m0 + v + 8 * g) * H_ + n] = acc[t].f[v] + bv;
  }
}

// ---------- kernel 6: residual + LayerNorm ----------

__global__ void __launch_bounds__(256) k_ln(const float* __restrict__ dense,
                                            const float* __restrict__ hidden,
                                            const float* __restrict__ gamma,
                                            const float* __restrict__ beta,
                                            float* __restrict__ out) {
  __shared__ float red[8];
  int row = blockIdx.x;
  int t = threadIdx.x;
  const float* dr = dense + (size_t)row * H_;
  const float* hr = hidden + (size_t)row * H_;
  float x[4], s = 0.f;
#pragma unroll
  for (int i = 0; i < 4; i++) {
    x[i] = dr[i * 256 + t] + hr[i * 256 + t];
    s += x[i];
  }
#pragma unroll
  for (int mm = 1; mm < 32; mm <<= 1) s += __shfl_xor(s, mm, 32);
  if ((t & 31) == 0) red[t >> 5] = s;
  __syncthreads();
  float tot = 0.f;
#pragma unroll
  for (int w = 0; w < 8; w++) tot += red[w];
  float mu = tot * (1.0f / H_);

  float vs = 0.f;
#pragma unroll
  for (int i = 0; i < 4; i++) {
    float d = x[i] - mu;
    vs += d * d;
  }
#pragma unroll
  for (int mm = 1; mm < 32; mm <<= 1) vs += __shfl_xor(vs, mm, 32);
  __syncthreads();
  if ((t & 31) == 0) red[t >> 5] = vs;
  __syncthreads();
  tot = 0.f;
#pragma unroll
  for (int w = 0; w < 8; w++) tot += red[w];
  float inv = rsqrtf(tot * (1.0f / H_) + 1e-5f);
#pragma unroll
  for (int i = 0; i < 4; i++) {
    int c = i * 256 + t;
    out[(size_t)row * H_ + c] = (x[i] - mu) * inv * gamma[c] + beta[c];
  }
}

// ---------- host launcher ----------

extern "C" void kernel_launch(void* const* d_in, const int* in_sizes, int n_in,
                              void* d_out, int out_size, void* d_ws, size_t ws_size,
                              hipStream_t stream) {
  const float* hidden = (const float*)d_in[0];
  const float* mask   = (const float*)d_in[1];
  const float* Wqkv   = (const float*)d_in[2];
  const float* bqkv   = (const float*)d_in[3];
  const float* Wd     = (const float*)d_in[4];
  const float* bd     = (const float*)d_in[5];
  const float* gamma  = (const float*)d_in[6];
  const float* beta   = (const float*)d_in[7];
  float* out = (float*)d_out;

  char* ws = (char*)d_ws;
  unsigned short* Xb  = (unsigned short*)ws; ws += (size_t)BS_ * H_ * 2;       // 16 MB
  unsigned short* Wqt = (unsigned short*)ws; ws += (size_t)NQKV_ * H_ * 2;     // 6 MB
  unsigned short* Wdt = (unsigned short*)ws; ws += (size_t)H_ * H_ * 2;        // 2 MB
  size_t headBytes = (size_t)B_ * NH_ * S_ * D_ * 2;                           // 16 MB each
  unsigned short* Qb = (unsigned short*)ws; ws += headBytes;
  unsigned short* Kb = (unsigned short*)ws; ws += headBytes;
  unsigned short* Vb = (unsigned short*)ws; ws += headBytes;
  unsigned short* Cb = (unsigned short*)ws; ws += (size_t)BS_ * H_ * 2;        // 16 MB
  float* Dn = (float*)ws;                                                      // 32 MB

  k_f32_to_bf16<<<(BS_ * H_) / 1024, 256, 0, stream>>>(hidden, Xb, BS_ * H_);

  dim3 g1(NQKV_ / 32, H_ / 32);
  k_transpose_bf16<<<g1, 256, 0, stream>>>(Wqkv, Wqt, H_, NQKV_);
  dim3 g2(H_ / 32, H_ / 32);
  k_transpose_bf16<<<g2, 256, 0, stream>>>(Wd, Wdt, H_, H_);

  k_gemm_qkv<<<(BS_ / 128) * (NQKV_ / 64), 256, 0, stream>>>(Xb, Wqt, bqkv, Qb, Kb, Vb);

  k_attention<<<B_ * NH_ * (S_ / 128), 256, 0, stream>>>(Qb, Kb, Vb, mask, Cb);

  k_gemm_dense<<<(BS_ / 128) * (H_ / 64), 256, 0, stream>>>(Cb, Wdt, bd, Dn);

  k_ln<<<BS_, 256, 0, stream>>>(Dn, hidden, gamma, beta, out);
}
